// NestedEventMamba_67740224192688
// MI455X (gfx1250) — matrix operand order repaired and checked by reading, not compiled, not run on hardware
//
#include <hip/hip_runtime.h>
#include <math.h>

// ---------------------------------------------------------------------------
// NestedEventMamba forward for MI455X (gfx1250, wave32, WMMA).
// All dense layers run through v_wmma_f32_16x16x32_f16 (f32 accum, f16
// operands converted in-register from f32 held in L2). Fragment loads are
// unconditional global_load_b128: host pads all GEMMs to K%32==0 and
// 16B-aligned row strides; M/N edges handled by row-clamping (garbage only
// reaches store-masked D entries). Selective scan: 1 WG / sequence,
// 1 thread / channel, 32-deep state in VGPRs, B/C staged via LDS.
// ---------------------------------------------------------------------------

typedef _Float16 half_t;
typedef __attribute__((ext_vector_type(16))) _Float16 v16h;
typedef __attribute__((ext_vector_type(8)))  float    v8f;

#define ACT_NONE     0
#define ACT_GELU     1
#define ACT_SILU     2
#define ACT_SOFTPLUS 3

#define RMS_EPS 1.1920929e-07f
#define LN_EPS  1e-05f

static const int cB = 8, cS = 16, cL = 256, cH = 128;
static const size_t N1 = (size_t)cB * cS * cL;   // 32768 intra rows
#define FEAT_LD 160                              // 130 padded to 32-multiple
#define XDBL_LD 96                               // [dt 0:8 | zeros 8:32 | B 32:64 | C 64:96]

// ------------------------------ device math --------------------------------
__device__ __forceinline__ float siluf(float x) { return x / (1.f + __expf(-x)); }
__device__ __forceinline__ float geluf(float x) { return 0.5f * x * (1.f + erff(x * 0.70710678118654752f)); }
__device__ __forceinline__ float softplusf(float x) { return x > 20.f ? x : log1pf(__expf(x)); }
__device__ __forceinline__ float act_apply(float x, int a) {
  if (a == ACT_GELU) return geluf(x);
  if (a == ACT_SILU) return siluf(x);
  if (a == ACT_SOFTPLUS) return softplusf(x);
  return x;
}

// ------------------------------ WMMA GEMM ----------------------------------
// C[M,N] = act(A[M,K via lda] @ W[N,K via ldw]^T + bias)
// REQUIRES (host-enforced): K%32==0, lda%4==0, ldw%4==0, 16B-aligned bases.
// One wave -> one 16x16 tile. Per K-step: 8x global_load_b128 + cvt + 1 WMMA.
__global__ void gemm_wmma(const float* __restrict__ A, int lda,
                          const float* __restrict__ W, int ldw,
                          const float* __restrict__ bias,
                          float* __restrict__ C, int ldc,
                          int M, int N, int K, int act)
{
  int wave = (int)((blockIdx.x * (size_t)blockDim.x + threadIdx.x) >> 5);
  int lane = threadIdx.x & 31;
  int mtiles = (M + 15) >> 4;
  int ntiles = (N + 15) >> 4;
  int mt = wave / ntiles;
  int nt = wave - mt * ntiles;
  if (mt >= mtiles) return;               // wave-uniform exit (EXEC stays full)

  int m0 = mt << 4, n0 = nt << 4;
  int hsel = lane >> 4;                   // half-lane select (0/1)
  int mr = m0 + (lane & 15); mr = (mr < M) ? mr : (M - 1);   // clamp: garbage
  int nr = n0 + (lane & 15); nr = (nr < N) ? nr : (N - 1);   // lands masked
  const float* Ap = A + (size_t)mr * lda + 8 * hsel;
  const float* Wp = W + (size_t)nr * ldw + 8 * hsel;

  v8f acc = {};
  for (int k0 = 0; k0 < K; k0 += 32) {
    __builtin_prefetch(Ap + k0 + 64, 0, 1);   // global_prefetch_b8
    // A fragment: af[0..7] = A[mr][k0+8h .. +7], af[8..15] = A[mr][k0+16+8h ..]
    float4 a0 = *(const float4*)(Ap + k0);
    float4 a1 = *(const float4*)(Ap + k0 + 4);
    float4 a2 = *(const float4*)(Ap + k0 + 16);
    float4 a3 = *(const float4*)(Ap + k0 + 20);
    // B fragment: bf[0..7] = W[nr][k0+8h .. +7], bf[8..15] = W[nr][k0+16+8h ..]
    float4 b0 = *(const float4*)(Wp + k0);
    float4 b1 = *(const float4*)(Wp + k0 + 4);
    float4 b2 = *(const float4*)(Wp + k0 + 16);
    float4 b3 = *(const float4*)(Wp + k0 + 20);
    float av[16] = {a0.x,a0.y,a0.z,a0.w, a1.x,a1.y,a1.z,a1.w,
                    a2.x,a2.y,a2.z,a2.w, a3.x,a3.y,a3.z,a3.w};
    float bv[16] = {b0.x,b0.y,b0.z,b0.w, b1.x,b1.y,b1.z,b1.w,
                    b2.x,b2.y,b2.z,b2.w, b3.x,b3.y,b3.z,b3.w};
    v16h af, bf;
    #pragma unroll
    for (int i = 0; i < 16; ++i) { af[i] = (half_t)av[i]; bf[i] = (half_t)bv[i]; }
    acc = __builtin_amdgcn_wmma_f32_16x16x32_f16(false, af, false, bf,
                                                 (short)0, acc, false, false);
  }

  int ccol = n0 + (lane & 15);
  if (ccol < N) {
    float badd = bias ? bias[ccol] : 0.f;
    #pragma unroll
    for (int r = 0; r < 8; ++r) {
      int crow = m0 + r + 8 * hsel;      // C/D layout: M = r + 8*(lane/16)
      if (crow < M)
        C[(size_t)crow * ldc + ccol] = act_apply(acc[r] + badd, act);
    }
  }
}

// --------------------------- weight pad staging ----------------------------
// adapter w1 (128x130) -> (128x160) zero-padded
__global__ void pad_adw1(const float* __restrict__ src, float* __restrict__ dst)
{
  int idx = blockIdx.x * 256 + threadIdx.x;
  if (idx >= 128 * FEAT_LD) return;
  int r = idx / FEAT_LD, c = idx % FEAT_LD;
  dst[idx] = (c < 130) ? src[r * 130 + c] : 0.f;
}
// x_proj w (72x256) -> (96x256): rows 0:8 dt, 8:32 zero, 32:96 = src rows 8:72
__global__ void pad_xpw(const float* __restrict__ src, float* __restrict__ dst)
{
  int idx = blockIdx.x * 256 + threadIdx.x;
  if (idx >= XDBL_LD * 256) return;
  int r = idx >> 8, c = idx & 255;
  float v = 0.f;
  if (r < 8)        v = src[r * 256 + c];
  else if (r >= 32) v = src[(r - 24) * 256 + c];
  dst[idx] = v;
}
// dt w (256x8) -> (256x32) zero-padded
__global__ void pad_dtw(const float* __restrict__ src, float* __restrict__ dst)
{
  int idx = blockIdx.x * 256 + threadIdx.x;
  if (idx >= 256 * 32) return;
  int r = idx >> 5, c = idx & 31;
  dst[idx] = (c < 8) ? src[r * 8 + c] : 0.f;
}

// ------------------------- normalization (row-wise) ------------------------
// blockDim == D (128 or 256); grid = rows. In-place safe.
__global__ void rows_norm(const float* __restrict__ in, float* __restrict__ out,
                          const float* __restrict__ w, const float* __restrict__ b,
                          int D, int is_ln, int act, float eps)
{
  __shared__ float s1[256];
  __shared__ float s2[256];
  size_t row = blockIdx.x;
  int c = threadIdx.x;
  float x = in[row * (size_t)D + c];
  s1[c] = x; s2[c] = x * x;
  __syncthreads();
  for (int off = D >> 1; off > 0; off >>= 1) {
    if (c < off) { s1[c] += s1[c + off]; s2[c] += s2[c + off]; }
    __syncthreads();
  }
  float mean = s1[0] / D, msq = s2[0] / D;
  float y;
  if (is_ln) {
    float var = msq - mean * mean;
    y = (x - mean) * rsqrtf(var + eps) * w[c] + b[c];
  } else {
    y = x * rsqrtf(msq + eps) * w[c];
  }
  out[row * (size_t)D + c] = act_apply(y, act);
}

// ----------------------------- conv kernels --------------------------------
__global__ void block_conv(const float* __restrict__ in, const float* __restrict__ w,
                           const float* __restrict__ b, float* __restrict__ out,
                           int L, int C, int k, int pad, size_t n)
{
  size_t idx = blockIdx.x * (size_t)blockDim.x + threadIdx.x;
  if (idx >= n) return;
  int c = (int)(idx % C);
  size_t rl = idx / C;
  int l = (int)(rl % L);
  size_t seq = rl / L;
  float acc = b[c];
  for (int j = 0; j < k; ++j) {
    int q = l - pad + j;
    if (q >= 0 && q < L)
      acc += w[c * k + j] * in[(seq * (size_t)L + q) * C + c];
  }
  out[idx] = acc;
}

// mamba causal depthwise conv (k=4, left pad 3) + bias + silu; direction-aware
__global__ void mamba_conv(const float* __restrict__ xz, const float* __restrict__ w,
                           const float* __restrict__ b, float* __restrict__ xm,
                           int L, int dir, size_t n)
{
  size_t idx = blockIdx.x * (size_t)blockDim.x + threadIdx.x;
  if (idx >= n) return;
  int c = (int)(idx & 255);
  size_t rl = idx >> 8;
  int p = (int)(rl % L);
  size_t seq = rl / L;
  float acc = b[c];
  #pragma unroll
  for (int j = 0; j < 4; ++j) {
    int q = p - 3 + j;
    if (q >= 0) {
      int phys = dir ? (L - 1 - q) : q;
      acc += w[c * 4 + j] * xz[(seq * (size_t)L + phys) * 512 + c];
    }
  }
  xm[idx] = siluf(acc);
}

// grouped post conv: groups=128, 2 in-ch/group, k=3, pad 1, +bias, silu
__global__ void post_conv(const float* __restrict__ xsn, const float* __restrict__ w,
                          const float* __restrict__ b, float* __restrict__ out,
                          int L, size_t n)
{
  size_t idx = blockIdx.x * (size_t)blockDim.x + threadIdx.x;
  if (idx >= n) return;
  int c = (int)(idx & 127);
  size_t rl = idx >> 7;
  int l = (int)(rl % L);
  size_t seq = rl / L;
  float acc = b[c];
  #pragma unroll
  for (int t = 0; t < 3; ++t) {
    int q = l - 1 + t;
    if (q >= 0 && q < L) {
      const float* r = xsn + (seq * (size_t)L + q) * 256;
      acc += w[(c * 2 + 0) * 3 + t] * r[2 * c]
           + w[(c * 2 + 1) * 3 + t] * r[2 * c + 1];
    }
  }
  out[idx] = siluf(acc);
}

// ----------------------------- selective scan ------------------------------
// grid = nseq, block = 256 (di channels). 32-state per thread in VGPRs.
__global__ void scan_kernel(const float* __restrict__ dt, const float* __restrict__ xdbl,
                            const float* __restrict__ xm, const float* __restrict__ A_log,
                            float* __restrict__ ys, int L)
{
  int seq = blockIdx.x;
  int c = threadIdx.x;
  __shared__ float sBC[64];   // [0:32) = B row, [32:64) = C row
  float A[32], h[32];
  #pragma unroll
  for (int s = 0; s < 32; ++s) { A[s] = -__expf(A_log[c * 32 + s]); h[s] = 0.f; }
  for (int t = 0; t < L; ++t) {
    size_t row = (size_t)seq * L + t;
    if (c < 64) sBC[c] = xdbl[row * XDBL_LD + 32 + c];   // B|C contiguous
    __syncthreads();
    float d  = dt[row * 256 + c];
    float du = d * xm[row * 256 + c];
    float y = 0.f;
    #pragma unroll
    for (int s = 0; s < 32; ++s) {
      h[s] = __expf(d * A[s]) * h[s] + du * sBC[s];
      y += h[s] * sBC[32 + s];
    }
    ys[row * 256 + c] = y;
    __syncthreads();
  }
}

// ----------------------------- elementwise ---------------------------------
__global__ void ew_y(float* __restrict__ ys, const float* __restrict__ xm,
                     const float* __restrict__ xz, const float* __restrict__ Dp,
                     int L, int dir, size_t n)
{
  size_t idx = blockIdx.x * (size_t)blockDim.x + threadIdx.x;
  if (idx >= n) return;
  int c = (int)(idx & 255);
  size_t rl = idx >> 8;
  int p = (int)(rl % L);
  size_t seq = rl / L;
  int phys = dir ? (L - 1 - p) : p;
  float z = xz[(seq * (size_t)L + phys) * 512 + 256 + c];
  ys[idx] = (ys[idx] + xm[idx] * Dp[c]) * siluf(z);
}

__global__ void ew_xs(const float* __restrict__ outd, const float* __restrict__ xn,
                      float* __restrict__ xs, int L, int dir, size_t n)
{
  size_t idx = blockIdx.x * (size_t)blockDim.x + threadIdx.x;
  if (idx >= n) return;
  int c = (int)(idx & 127);
  size_t rl = idx >> 7;
  int p = (int)(rl % L);
  size_t seq = rl / L;
  size_t phys = dir ? (seq * (size_t)L + (L - 1 - p)) : rl;
  xs[phys * 256 + (dir ? 128 : 0) + c] = outd[rl * 128 + c] + xn[phys * 128 + c];
}

__global__ void ew_combine(float* __restrict__ x, const float* __restrict__ xmix,
                           const float* __restrict__ xmlp, const float* __restrict__ bl,
                           size_t n)
{
  size_t idx = blockIdx.x * (size_t)blockDim.x + threadIdx.x;
  if (idx >= n) return;
  float e0 = __expf(bl[0]), e1 = __expf(bl[1]);
  float w0 = e0 / (e0 + e1);
  x[idx] += w0 * xmix[idx] + (1.f - w0) * xmlp[idx];
}

__global__ void ew_mask(const float* __restrict__ wv, const int* __restrict__ nseg,
                        float* __restrict__ seqo, size_t n)
{
  size_t idx = blockIdx.x * (size_t)blockDim.x + threadIdx.x;
  if (idx >= n) return;
  size_t r = idx >> 7;
  int b = (int)(r >> 4), s = (int)(r & 15);
  seqo[idx] = (s < nseg[b]) ? wv[idx] : 0.f;
}

// feats = concat(z_real, z_imag, coords, zero-pad) -> (rows, FEAT_LD)
__global__ void build_feats(const float* __restrict__ zr, const float* __restrict__ zi,
                            const float* __restrict__ co, float* __restrict__ feats,
                            size_t n)
{
  size_t idx = blockIdx.x * (size_t)blockDim.x + threadIdx.x;
  if (idx >= n) return;
  size_t r = idx / FEAT_LD;
  int j = (int)(idx % FEAT_LD);
  float v = 0.f;
  if (j < 64)       v = zr[r * 64 + j];
  else if (j < 128) v = zi[r * 64 + (j - 64)];
  else if (j < 130) v = co[r * 2 + (j - 128)];
  feats[idx] = v;
}

// ------------------------------- pooling -----------------------------------
__global__ void pool_logits(const float* __restrict__ h, const float* __restrict__ w2,
                            const float* __restrict__ b2, float* __restrict__ lg,
                            int rows, int K)
{
  int gw = (int)((blockIdx.x * (size_t)blockDim.x + threadIdx.x) >> 5);
  int lane = threadIdx.x & 31;
  if (gw >= rows) return;
  const float* hr = h + (size_t)gw * K;
  float s = 0.f;
  for (int k = lane; k < K; k += 32) s += hr[k] * w2[k];
  #pragma unroll
  for (int off = 16; off > 0; off >>= 1) s += __shfl_down(s, off, 32);
  if (lane == 0) lg[gw] = s + b2[0];
}

__global__ void pool_reduce(const float* __restrict__ x, const float* __restrict__ lg,
                            const int* __restrict__ lens, float* __restrict__ out,
                            int L, int D)
{
  __shared__ float sh[256];
  __shared__ float wgt[256];
  int seq = blockIdx.x, t = threadIdx.x;
  int len = lens[seq];
  float e = (t < L && t < len) ? lg[(size_t)seq * L + t] : -INFINITY;
  sh[t] = e; __syncthreads();
  for (int off = 128; off > 0; off >>= 1) {
    if (t < off) sh[t] = fmaxf(sh[t], sh[t + off]);
    __syncthreads();
  }
  float mx = sh[0]; __syncthreads();
  float ex = (e > -INFINITY) ? __expf(e - mx) : 0.f;
  sh[t] = ex; __syncthreads();
  for (int off = 128; off > 0; off >>= 1) {
    if (t < off) sh[t] += sh[t + off];
    __syncthreads();
  }
  float inv = 1.f / sh[0];
  wgt[t] = ex * inv; __syncthreads();
  if (t < D) {
    float acc = 0.f;
    for (int l = 0; l < L; ++l)
      acc += x[((size_t)seq * L + l) * D + t] * wgt[l];
    out[(size_t)seq * D + t] = acc;
  }
}

// ----------------------------- host plumbing -------------------------------
struct Blk {
  const float *branch, *conv_b, *conv_w, *ln_conv_b, *ln_conv_w,
              *ln_post_b, *ln_post_w, *mlp_b1, *mlp_b2, *mlp_w1, *mlp_w2,
              *norm_w, *post_conv_b, *post_conv_w,
              *A_log, *Dp, *sconv_b, *sconv_w, *dt_b, *dt_w, *in_w, *out_w, *xp_w;
};

static Blk get_blk(void* const* d_in, int base) {
  Blk b;
  const float** p = (const float**)&b;
  for (int i = 0; i < 23; ++i) p[i] = (const float*)d_in[base + i];
  return b;
}

static void gemm(hipStream_t st, const float* A, int lda, const float* W, int ldw,
                 const float* bias, float* C, int ldc, int M, int N, int K, int act)
{
  int mt = (M + 15) / 16, nt = (N + 15) / 16;
  long long waves = (long long)mt * nt;
  int blocks = (int)((waves + 7) / 8);
  gemm_wmma<<<blocks, 256, 0, st>>>(A, lda, W, ldw, bias, C, ldc, M, N, K, act);
}

static inline int div_up(size_t n, int b) { return (int)((n + b - 1) / b); }

static void run_block(hipStream_t st, const Blk& bp, float* X,
                      int rows, int nseq, int L, int k,
                      float* XN, float* XC, float* XZ, float* XM,
                      float* XDBL, float* DT, float* YS, float* OUTD, float* XS,
                      float* XPWpad, float* DTWpad)
{
  int pad = (k - 1) / 2;
  size_t n128 = (size_t)rows * 128, n256 = (size_t)rows * 256;

  // stage padded mamba weights for this block
  pad_xpw<<<div_up((size_t)XDBL_LD * 256, 256), 256, 0, st>>>(bp.xp_w, XPWpad);
  pad_dtw<<<div_up((size_t)256 * 32, 256), 256, 0, st>>>(bp.dt_w, DTWpad);

  rows_norm<<<rows, 128, 0, st>>>(X, XN, bp.norm_w, nullptr, 128, 0, ACT_NONE, RMS_EPS);
  block_conv<<<div_up(n128, 256), 256, 0, st>>>(XN, bp.conv_w, bp.conv_b, XC, L, 128, k, pad, n128);
  rows_norm<<<rows, 128, 0, st>>>(XC, XC, bp.ln_conv_w, bp.ln_conv_b, 128, 1, ACT_SILU, LN_EPS);
  gemm(st, XC, 128, bp.in_w, 128, nullptr, XZ, 512, rows, 512, 128, ACT_NONE);

  for (int dir = 0; dir < 2; ++dir) {
    mamba_conv<<<div_up(n256, 256), 256, 0, st>>>(XZ, bp.sconv_w, bp.sconv_b, XM, L, dir, n256);
    gemm(st, XM, 256, XPWpad, 256, nullptr, XDBL, XDBL_LD, rows, XDBL_LD, 256, ACT_NONE);
    gemm(st, XDBL, XDBL_LD, DTWpad, 32, bp.dt_b, DT, 256, rows, 256, 32, ACT_SOFTPLUS);
    scan_kernel<<<nseq, 256, 0, st>>>(DT, XDBL, XM, bp.A_log, YS, L);
    ew_y<<<div_up(n256, 256), 256, 0, st>>>(YS, XM, XZ, bp.Dp, L, dir, n256);
    gemm(st, YS, 256, bp.out_w, 256, nullptr, OUTD, 128, rows, 128, 256, ACT_NONE);
    ew_xs<<<div_up(n128, 256), 256, 0, st>>>(OUTD, XN, XS, L, dir, n128);
  }

  rows_norm<<<rows, 256, 0, st>>>(XS, XS, bp.ln_post_w, bp.ln_post_b, 256, 1, ACT_NONE, LN_EPS);
  post_conv<<<div_up(n128, 256), 256, 0, st>>>(XS, bp.post_conv_w, bp.post_conv_b, XC /*=XMIX*/, L, n128);
  gemm(st, XN, 128, bp.mlp_w1, 128, bp.mlp_b1, XZ /*=MLPH*/, 512, rows, 512, 128, ACT_GELU);
  gemm(st, XZ, 512, bp.mlp_w2, 512, bp.mlp_b2, OUTD /*=XMLP*/, 128, rows, 128, 512, ACT_NONE);
  ew_combine<<<div_up(n128, 256), 256, 0, st>>>(X, XC, OUTD, bp.branch, n128);
}

// d_in layout: params pytree leaves (sorted dict keys at every level) first,
// then z_real, z_imag, coords, seg_lengths, num_segments.
enum {
  I_AD_B1 = 0, I_AD_B2, I_AD_W1, I_AD_W2,
  I_HD_B1, I_HD_B2, I_HD_LNB, I_HD_LNW, I_HD_W1, I_HD_W2,
  I_INTER_B0 = 10, I_INTER_B1 = 33,
  I_INTER_POOL = 56,      // b1, b2, w1, w2
  I_INTRA_B0 = 60, I_INTRA_B1 = 83,
  I_INTRA_POOL = 106,
  I_ZREAL = 110, I_ZIMAG, I_COORDS, I_SEGLEN, I_NUMSEG
};

extern "C" void kernel_launch(void* const* d_in, const int* in_sizes, int n_in,
                              void* d_out, int out_size, void* d_ws, size_t ws_size,
                              hipStream_t stream)
{
  (void)in_sizes; (void)n_in; (void)out_size; (void)ws_size;

  const float* zr = (const float*)d_in[I_ZREAL];
  const float* zi = (const float*)d_in[I_ZIMAG];
  const float* co = (const float*)d_in[I_COORDS];
  const int* seg_len = (const int*)d_in[I_SEGLEN];
  const int* num_seg = (const int*)d_in[I_NUMSEG];

  const float* ad_b1 = (const float*)d_in[I_AD_B1];
  const float* ad_b2 = (const float*)d_in[I_AD_B2];
  const float* ad_w1 = (const float*)d_in[I_AD_W1];
  const float* ad_w2 = (const float*)d_in[I_AD_W2];
  const float* hd_b1 = (const float*)d_in[I_HD_B1];
  const float* hd_b2 = (const float*)d_in[I_HD_B2];
  const float* hd_lnb = (const float*)d_in[I_HD_LNB];
  const float* hd_lnw = (const float*)d_in[I_HD_LNW];
  const float* hd_w1 = (const float*)d_in[I_HD_W1];
  const float* hd_w2 = (const float*)d_in[I_HD_W2];
  const float* ip_b1 = (const float*)d_in[I_INTRA_POOL + 0];
  const float* ip_b2 = (const float*)d_in[I_INTRA_POOL + 1];
  const float* ip_w1 = (const float*)d_in[I_INTRA_POOL + 2];
  const float* ip_w2 = (const float*)d_in[I_INTRA_POOL + 3];
  const float* ep_b1 = (const float*)d_in[I_INTER_POOL + 0];
  const float* ep_b2 = (const float*)d_in[I_INTER_POOL + 1];
  const float* ep_w1 = (const float*)d_in[I_INTER_POOL + 2];
  const float* ep_w2 = (const float*)d_in[I_INTER_POOL + 3];

  Blk ia0 = get_blk(d_in, I_INTRA_B0), ia1 = get_blk(d_in, I_INTRA_B1);
  Blk ie0 = get_blk(d_in, I_INTER_B0), ie1 = get_blk(d_in, I_INTER_B1);

  // workspace carve (floats)
  float* ws = (float*)d_ws;
  size_t off = 0;
  auto carve = [&](size_t nf) { float* p = ws + off; off += nf; return p; };
  float* X    = carve(N1 * 128);
  float* XN   = carve(N1 * 128);
  float* XC   = carve(N1 * 128);        // also XMIX
  float* XZ   = carve(N1 * 512);        // also FEATS (N1*FEAT_LD) and MLP hidden
  float* XM   = carve(N1 * 256);        // also pool hidden
  float* XDBL = carve(N1 * XDBL_LD);    // also pool logits (N1)
  float* DT   = carve(N1 * 256);
  float* YS   = carve(N1 * 256);
  float* OUTD = carve(N1 * 128);        // also XMLP
  float* XS   = carve(N1 * 256);        // normalized in place
  float* WV   = carve(128 * 128);
  float* SEQ  = carve(128 * 128);
  float* G    = carve(8 * 128);
  float* G2   = carve(8 * 128);
  float* HH   = carve(8 * 64);
  float* ADW1p = carve(128 * FEAT_LD);  // padded adapter w1
  float* XPWp  = carve(XDBL_LD * 256);  // padded x_proj w (per block, restaged)
  float* DTWp  = carve(256 * 32);       // padded dt w (per block, restaged)

  // ---- adapter: feats -> gelu(w1) -> w2 ----
  size_t nfeat = N1 * FEAT_LD;
  build_feats<<<div_up(nfeat, 256), 256, 0, stream>>>(zr, zi, co, XZ, nfeat);
  pad_adw1<<<div_up((size_t)128 * FEAT_LD, 256), 256, 0, stream>>>(ad_w1, ADW1p);
  gemm(stream, XZ, FEAT_LD, ADW1p, FEAT_LD, ad_b1, XN, 128, (int)N1, 128, FEAT_LD, ACT_GELU);
  gemm(stream, XN, 128, ad_w2, 128, ad_b2, X, 128, (int)N1, 128, 128, ACT_NONE);

  // ---- intra blocks: 128 sequences of L=256, k=11 ----
  run_block(stream, ia0, X, (int)N1, 128, cL, 11, XN, XC, XZ, XM, XDBL, DT, YS, OUTD, XS, XPWp, DTWp);
  run_block(stream, ia1, X, (int)N1, 128, cL, 11, XN, XC, XZ, XM, XDBL, DT, YS, OUTD, XS, XPWp, DTWp);

  // ---- intra pooling over L ----
  gemm(stream, X, 128, ip_w1, 128, ip_b1, XM, 256, (int)N1, 256, 128, ACT_GELU);
  pool_logits<<<div_up((size_t)N1 * 32, 256), 256, 0, stream>>>(XM, ip_w2, ip_b2, XDBL, (int)N1, 256);
  pool_reduce<<<128, 256, 0, stream>>>(X, XDBL, seg_len, WV, cL, 128);
  ew_mask<<<div_up((size_t)128 * 128, 256), 256, 0, stream>>>(WV, num_seg, SEQ, (size_t)128 * 128);

  // ---- inter blocks: 8 sequences of L=16, k=3 ----
  run_block(stream, ie0, SEQ, 128, cB, cS, 3, XN, XC, XZ, XM, XDBL, DT, YS, OUTD, XS, XPWp, DTWp);
  run_block(stream, ie1, SEQ, 128, cB, cS, 3, XN, XC, XZ, XM, XDBL, DT, YS, OUTD, XS, XPWp, DTWp);

  // ---- inter pooling over S ----
  gemm(stream, SEQ, 128, ep_w1, 128, ep_b1, XM, 256, 128, 256, 128, ACT_GELU);
  pool_logits<<<div_up((size_t)128 * 32, 256), 256, 0, stream>>>(XM, ep_w2, ep_b2, XDBL, 128, 256);
  pool_reduce<<<cB, 256, 0, stream>>>(SEQ, XDBL, num_seg, G, cS, 128);

  // ---- head ----
  rows_norm<<<cB, 128, 0, stream>>>(G, G2, hd_lnw, hd_lnb, 128, 1, ACT_NONE, LN_EPS);
  gemm(stream, G2, 128, hd_w1, 128, hd_b1, HH, 64, cB, 64, 128, ACT_GELU);
  gemm(stream, HH, 64, hd_w2, 64, hd_b2, (float*)d_out, 11, cB, 11, 64, ACT_NONE);
}
